// RelPartialLearnableMultiHeadAttn_41300405518612
// MI455X (gfx1250) — compile-verified, matrix operand order checked by reading, not run on hardware
//
#include <hip/hip_runtime.h>

typedef __bf16 bf16;
typedef unsigned int u32;
typedef __attribute__((ext_vector_type(4)))  bf16  v4bf;
typedef __attribute__((ext_vector_type(8)))  bf16  v8bf;
typedef __attribute__((ext_vector_type(16))) bf16  v16bf;
typedef __attribute__((ext_vector_type(8)))  float v8f;
typedef __attribute__((ext_vector_type(4)))  u32   v4u;
typedef __attribute__((ext_vector_type(8)))  int   v8i;
typedef __attribute__((ext_vector_type(4)))  int   v4i;

#define QLEN 1024
#define MLEN 1024
#define KLEN 2048
#define BSZ  4
#define DM   1024
#define NH   16
#define DH   64

#if __has_builtin(__builtin_amdgcn_tensor_load_to_lds) && \
    __has_builtin(__builtin_amdgcn_s_wait_tensorcnt)
#define USE_TDM 1
#else
#define USE_TDM 0
#endif

// ---------------------------------------------------------------------------
// WMMA helpers (CDNA5 wave32, V_WMMA_F32_16X16X32_BF16)
// A 16x32 frag: lane (m = lane&15, half = lane>>4) holds k in
//   {half*8 .. half*8+7} u {16+half*8 .. 16+half*8+7}  -> two 16B runs.
// B 32x16 frag mirrors it with n = lane&15.  C frag: vgpr v, lanes 0-15 ->
// M=v, lanes 16-31 -> M=v+8; N=lane&15.
// ---------------------------------------------------------------------------
__device__ __forceinline__ v16bf cat8(v8bf lo, v8bf hi) {
  return __builtin_shufflevector(lo, hi, 0,1,2,3,4,5,6,7,8,9,10,11,12,13,14,15);
}
__device__ __forceinline__ v16bf frag_ld(const bf16* rowp, int half) {
  v8bf lo = *(const v8bf*)(rowp + half * 8);
  v8bf hi = *(const v8bf*)(rowp + 16 + half * 8);
  return cat8(lo, hi);
}
__device__ __forceinline__ v8f wmma_bf16(v16bf a, v16bf b, v8f c) {
  return __builtin_amdgcn_wmma_f32_16x16x32_bf16(false, a, false, b, (short)0, c,
                                                 false, false);
}

#if USE_TDM
// ---------------------------------------------------------------------------
// 2D TDM tile load: global (row-major, 2-byte elems, row stride = stride0
// elements) -> LDS at lds_off, tile tl1 rows x tl0 cols, LDS padded +16B
// every 64B so the LDS row stride is 40 bf16 (matches frag_ld layout).
// D# packing per CDNA5 ISA (group0 128b, group1 256b; groups 2/3 zero).
// clang-23 builtin arity: (u32x4, i32x8, i32x4, i32x4, i32x8, i32 cpol).
// ---------------------------------------------------------------------------
__device__ __forceinline__ void tdm_load_2d(u32 lds_off, const void* gptr,
                                            u32 td0, u32 td1, u32 tl0, u32 tl1,
                                            unsigned long long stride0) {
  unsigned long long ga = (unsigned long long)gptr;
  v4u g0;
  g0.x = 1u;                                 // count=1 (valid user descriptor)
  g0.y = lds_off;                            // lds_addr (bytes)
  g0.z = (u32)ga;                            // global_addr[31:0]
  g0.w = (u32)(ga >> 32) | (2u << 30);       // global_addr[56:32] | type=2
  u32 d0 = (1u << 16)                        // data_size = 2 bytes
         | (1u << 20)                        // pad_enable
         | (3u << 22)                        // pad_interval: 16 DWORDs (64B)
         | (3u << 25);                       // pad_amount:   4 DWORDs (16B)
  u32 d1 = (td0 & 0xffffu) << 16;                       // tensor_dim0 lo
  u32 d2 = (td0 >> 16) | ((td1 & 0xffffu) << 16);       // dim0 hi | dim1 lo
  u32 d3 = (td1 >> 16) | (tl0 << 16);                   // dim1 hi | tile_dim0
  u32 d4 = tl1 & 0xffffu;                               // tile_dim1 (dim2=0)
  u32 d5 = (u32)stride0;                                // dim0_stride lo
  u32 d6 = (u32)(stride0 >> 32) & 0xffffu;              // stride hi (dim1_str=0)
  v8i g1 = { (int)d0, (int)d1, (int)d2, (int)d3, (int)d4, (int)d5, (int)d6, 0 };
  v4i gz4 = { 0, 0, 0, 0 };
  v8i gz8 = { 0, 0, 0, 0, 0, 0, 0, 0 };
  __builtin_amdgcn_tensor_load_to_lds(g0, g1, gz4, gz4, gz8, 0);
}
#endif

__device__ __forceinline__ u32 ldsoff(const void* p) {
  // LDS flat aperture: addr[31:0] is the LDS byte offset (ISA 10.2 aperture calc)
  return (u32)(unsigned long long)p;
}

// ---------------------------------------------------------------------------
// f32 -> bf16 conversion pass (one-shot, bandwidth-trivial)
// ---------------------------------------------------------------------------
__global__ __launch_bounds__(256) void cvt_kernel(const float* __restrict__ src,
                                                  bf16* __restrict__ dst) {
  long i = ((long)blockIdx.x * 256 + threadIdx.x) * 4;
  float4 v = *(const float4*)(src + i);
  v4bf t = { (bf16)v.x, (bf16)v.y, (bf16)v.z, (bf16)v.w };
  *(v4bf*)(dst + i) = t;
}

// ---------------------------------------------------------------------------
// NT GEMM on pre-converted bf16: C[m,n] = sum_k A[m,k]*B[n,k].
// Block 128x64, 8 waves (4x2), each wave 32x32 = 2x2 WMMA frags, K-step 32.
// Tiles land in LDS via double-buffered TDM (wave 0 issues, TENSORcnt waits),
// or manual b128 staging if the builtin is unavailable.
// mode 0: qkv epilogue (q f32 / k bf16 / v-transposed bf16)
// mode 1: rk bf16     mode 2: plain f32
// ---------------------------------------------------------------------------
__global__ __launch_bounds__(256) void gemm_bf16_kernel(
    const bf16* __restrict__ A, const bf16* __restrict__ Bm,
    int M, int N, int K, int mode,
    float* __restrict__ outF, bf16* __restrict__ outB0, bf16* __restrict__ outB1)
{
  __shared__ __align__(16) bf16 sA[2][128 * 40];
  __shared__ __align__(16) bf16 sB[2][64 * 40];

  const int tid  = threadIdx.x;
  const int wave = tid >> 5;
  const int lane = tid & 31;
  const int wm   = (wave >> 1) * 32;
  const int wn   = (wave & 1) * 32;
  const int bm   = blockIdx.x * 128;
  const int bn   = blockIdx.y * 64;
  const int mloc = lane & 15;
  const int half = lane >> 4;
  const int nsteps = K / 32;

  v8f c[2][2] = {};

#if USE_TDM
  if (wave == 0) {
    tdm_load_2d(ldsoff(&sA[0][0]), A + (long)bm * K, K, M, 32, 128, K);
    tdm_load_2d(ldsoff(&sB[0][0]), Bm + (long)bn * K, K, N, 32, 64, K);
  }
#endif

  for (int s = 0; s < nsteps; ++s) {
    const int cur = s & 1;
#if USE_TDM
    if (wave == 0) __builtin_amdgcn_s_wait_tensorcnt(0);
    __syncthreads();                       // release buffer `cur` to all waves
    if (wave == 0 && s + 1 < nsteps) {     // prefetch next K-slab into cur^1
      long k0n = (long)(s + 1) * 32;
      tdm_load_2d(ldsoff(&sA[cur ^ 1][0]), A + (long)bm * K + k0n, K, M, 32, 128, K);
      tdm_load_2d(ldsoff(&sB[cur ^ 1][0]), Bm + (long)bn * K + k0n, K, N, 32, 64, K);
    }
#else
    {
      int k0 = s * 32;
      #pragma unroll
      for (int q = 0; q < 2; ++q) {        // A: 128 rows x 4 chunks of 16B
        int ch = tid + q * 256;
        int row = ch >> 2, seg = ch & 3;
        *(v8bf*)&sA[cur][row * 40 + seg * 8] =
            *(const v8bf*)(A + (long)(bm + row) * K + k0 + seg * 8);
      }
      {                                    // B: 64 rows x 4 chunks of 16B
        int row = tid >> 2, seg = tid & 3;
        *(v8bf*)&sB[cur][row * 40 + seg * 8] =
            *(const v8bf*)(Bm + (long)(bn + row) * K + k0 + seg * 8);
      }
      __syncthreads();
    }
#endif

    v16bf af[2], bfr[2];
    #pragma unroll
    for (int i = 0; i < 2; ++i)
      af[i] = frag_ld(&sA[cur][(wm + i * 16 + mloc) * 40], half);
    #pragma unroll
    for (int j = 0; j < 2; ++j)
      bfr[j] = frag_ld(&sB[cur][(wn + j * 16 + mloc) * 40], half);
    #pragma unroll
    for (int i = 0; i < 2; ++i)
      #pragma unroll
      for (int j = 0; j < 2; ++j)
        c[i][j] = wmma_bf16(af[i], bfr[j], c[i][j]);
    __syncthreads();                       // reads done before buffer reuse
  }

  // epilogue
  #pragma unroll
  for (int i = 0; i < 2; ++i)
    #pragma unroll
    for (int j = 0; j < 2; ++j)
      #pragma unroll
      for (int v = 0; v < 8; ++v) {
        int gm = bm + wm + i * 16 + v + 8 * half;
        int gn = bn + wn + j * 16 + mloc;
        float val = c[i][j][v];
        if (mode == 0) {
          int jrow = gm >> 2, b = gm & 3;
          if (gn < 1024) {                 // q head: keep last QLEN rows (f32)
            if (jrow >= MLEN)
              outF[((long)(jrow - MLEN) * 4 + b) * 1024 + gn] = val;
          } else if (gn < 2048) {          // k head (bf16), [klen*bsz][1024]
            outB0[(long)gm * 1024 + (gn - 1024)] = (bf16)val;
          } else {                         // v head transposed: [b][n][d][klen]
            int cv = gn - 2048, hn = cv >> 6, d = cv & 63;
            outB1[((long)((b * NH + hn) * DH + d)) * KLEN + jrow] = (bf16)val;
          }
        } else if (mode == 1) {            // rk bf16: row*1024 + gn
          outB0[(long)gm * 1024 + gn] = (bf16)val;
        } else {                           // plain f32
          outF[(long)gm * 1024 + gn] = val;
        }
      }
}

// ---------------------------------------------------------------------------
// Attention: one wave per (16-query block, head n, batch b).
//   rel_shift identity: BD[i,j] = BD_raw[i, j + QLEN-1-i] on unmasked entries.
//   BD_raw row-block (16 x 2048 f32) via WMMA into dynamic LDS, then a
//   flash-style online-softmax loop: AC (WMMA) + shifted BD + mask ->
//   P (bf16 via LDS) -> PV (WMMA).  Output written as bf16.
// ---------------------------------------------------------------------------
__global__ __launch_bounds__(32) void attn_kernel(
    const float* __restrict__ head_q, const bf16* __restrict__ head_k,
    const bf16* __restrict__ vT, const bf16* __restrict__ rk,
    const float* __restrict__ rwb, const float* __restrict__ rrb,
    bf16* __restrict__ attn_vec)
{
  extern __shared__ char smem[];
  float* bd = (float*)smem;                       // 16 x 2048 f32 = 128 KB
  bf16*  pl = (bf16*)(smem + 16 * 2048 * 4);      // 16 x 32 bf16 P staging

  const int i0   = blockIdx.x * 16;
  const int n    = blockIdx.y;
  const int b    = blockIdx.z;
  const int lane = threadIdx.x & 31;
  const int m    = lane & 15;
  const int half = lane >> 4;
  const float scale = 0.125f;                     // 1/sqrt(DH)

  v16bf qw[2], qr[2];
  {
    const float* qrow = head_q + ((long)(i0 + m) * 4 + b) * 1024 + n * DH;
    #pragma unroll
    for (int f = 0; f < 2; ++f) {
      int d0 = f * 32;
      #pragma unroll
      for (int e = 0; e < 16; ++e) {
        int d = d0 + ((e < 8) ? (half * 8 + e) : (16 + half * 8 + (e - 8)));
        float qv = qrow[d];
        qw[f][e] = (bf16)(qv + rwb[n * DH + d]);
        qr[f][e] = (bf16)(qv + rrb[n * DH + d]);
      }
    }
  }

  // BD_raw[i, jr] for jr tiles referenced: jr >= 1008 - i0
  for (int t = 63 - (i0 >> 4); t < 128; ++t) {
    int jr0 = t * 16;
    const bf16* rkrow = rk + ((long)(jr0 + m) * NH + n) * DH;
    v8f acc = {};
    acc = wmma_bf16(qr[0], frag_ld(rkrow, half), acc);
    acc = wmma_bf16(qr[1], frag_ld(rkrow + 32, half), acc);
    #pragma unroll
    for (int v = 0; v < 8; ++v)
      bd[(v + 8 * half) * 2048 + jr0 + m] = acc[v];
  }
  __syncthreads();

  float mrow[8], lrow[8];
  #pragma unroll
  for (int v = 0; v < 8; ++v) { mrow[v] = -__builtin_inff(); lrow[v] = 0.0f; }
  v8f o[4] = {};

  const int jmax = i0 + 15 + MLEN;
  for (int j0 = 0; j0 <= jmax; j0 += 32) {
    float sc[2][8];
    float tmax[8];
    #pragma unroll
    for (int v = 0; v < 8; ++v) tmax[v] = -__builtin_inff();

    #pragma unroll
    for (int s = 0; s < 2; ++s) {
      int j0s = j0 + s * 16;
      const bf16* krow = head_k + (((long)(j0s + m) * 4 + b) * NH + n) * DH;
      v8f acc = {};
      acc = wmma_bf16(qw[0], frag_ld(krow, half), acc);
      acc = wmma_bf16(qw[1], frag_ld(krow + 32, half), acc);
      #pragma unroll
      for (int v = 0; v < 8; ++v) {
        int row = v + 8 * half;
        int i = i0 + row;
        int j = j0s + m;
        float x;
        if (j > i + MLEN) {
          x = -__builtin_inff();
        } else {
          int jr = j + (QLEN - 1) - i;
          x = (acc[v] + bd[row * 2048 + jr]) * scale;
        }
        sc[s][v] = x;
        tmax[v] = fmaxf(tmax[v], x);
      }
    }

    #pragma unroll
    for (int v = 0; v < 8; ++v) {
      float tm = tmax[v];
      tm = fmaxf(tm, __shfl_xor(tm, 1));
      tm = fmaxf(tm, __shfl_xor(tm, 2));
      tm = fmaxf(tm, __shfl_xor(tm, 4));
      tm = fmaxf(tm, __shfl_xor(tm, 8));
      float mnew = fmaxf(mrow[v], tm);
      float corr = __expf(mrow[v] - mnew);
      float p0 = __expf(sc[0][v] - mnew);
      float p1 = __expf(sc[1][v] - mnew);
      float rs = p0 + p1;
      rs += __shfl_xor(rs, 1);
      rs += __shfl_xor(rs, 2);
      rs += __shfl_xor(rs, 4);
      rs += __shfl_xor(rs, 8);
      lrow[v] = lrow[v] * corr + rs;
      mrow[v] = mnew;
      #pragma unroll
      for (int g = 0; g < 4; ++g) o[g][v] = o[g][v] * corr;
      int row = v + 8 * half;
      pl[row * 32 + m]      = (bf16)p0;
      pl[row * 32 + 16 + m] = (bf16)p1;
    }
    __syncthreads();

    v16bf pf;
    #pragma unroll
    for (int e = 0; e < 16; ++e) {
      int kk = (e < 8) ? (half * 8 + e) : (16 + half * 8 + (e - 8));
      pf[e] = pl[m * 32 + kk];
    }
    const bf16* vbase = vT + (long)((b * NH + n) * DH) * KLEN + j0;
    #pragma unroll
    for (int g = 0; g < 4; ++g) {
      const bf16* vrow = vbase + (long)(g * 16 + m) * KLEN;
      o[g] = wmma_bf16(pf, frag_ld(vrow, half), o[g]);
    }
    __syncthreads();
  }

  #pragma unroll
  for (int g = 0; g < 4; ++g)
    #pragma unroll
    for (int v = 0; v < 8; ++v) {
      int row = v + 8 * half;
      float val = o[g][v] / lrow[v];
      attn_vec[((long)(i0 + row) * 4 + b) * 1024 + n * DH + g * 16 + m] = (bf16)val;
    }
}

// ---------------------------------------------------------------------------
// Residual add + LayerNorm: one block per (i,b) row of 1024.
// ---------------------------------------------------------------------------
__global__ __launch_bounds__(256) void ln_kernel(
    const float* __restrict__ w, const float* __restrict__ ao,
    const float* __restrict__ g, const float* __restrict__ bta,
    float* __restrict__ out)
{
  __shared__ float red[256];
  const long row = blockIdx.x;
  const float* x1 = w + row * 1024;
  const float* x2 = ao + row * 1024;
  float vals[4];
  float s = 0.0f;
  #pragma unroll
  for (int t = 0; t < 4; ++t) {
    int i = threadIdx.x + t * 256;
    vals[t] = x1[i] + x2[i];
    s += vals[t];
  }
  red[threadIdx.x] = s;
  __syncthreads();
  for (int off = 128; off > 0; off >>= 1) {
    if (threadIdx.x < off) red[threadIdx.x] += red[threadIdx.x + off];
    __syncthreads();
  }
  float mu = red[0] * (1.0f / 1024.0f);
  __syncthreads();
  float vs = 0.0f;
  #pragma unroll
  for (int t = 0; t < 4; ++t) { float d = vals[t] - mu; vs += d * d; }
  red[threadIdx.x] = vs;
  __syncthreads();
  for (int off = 128; off > 0; off >>= 1) {
    if (threadIdx.x < off) red[threadIdx.x] += red[threadIdx.x + off];
    __syncthreads();
  }
  float rstd = rsqrtf(red[0] * (1.0f / 1024.0f) + 1e-5f);
  #pragma unroll
  for (int t = 0; t < 4; ++t) {
    int i = threadIdx.x + t * 256;
    out[row * 1024 + i] = (vals[t] - mu) * rstd * g[i] + bta[i];
  }
}

// ---------------------------------------------------------------------------
extern "C" void kernel_launch(void* const* d_in, const int* in_sizes, int n_in,
                              void* d_out, int out_size, void* d_ws, size_t ws_size,
                              hipStream_t stream) {
  (void)in_sizes; (void)n_in; (void)out_size; (void)ws_size;
  const float* w    = (const float*)d_in[0];
  const float* r    = (const float*)d_in[1];
  const float* rwb  = (const float*)d_in[2];
  const float* rrb  = (const float*)d_in[3];
  const float* mems = (const float*)d_in[4];
  // d_in[5] = attn_mask (computed analytically: j > i + MLEN)
  const float* qkvW = (const float*)d_in[6];
  const float* rW   = (const float*)d_in[7];
  const float* oW   = (const float*)d_in[8];
  const float* ln_g = (const float*)d_in[9];
  const float* ln_b = (const float*)d_in[10];

  char* ws = (char*)d_ws;
  float* head_q = (float*)(ws);                     // 16 MB
  bf16*  head_k = (bf16*) (ws + (16l << 20));       // 16 MB
  bf16*  vT     = (bf16*) (ws + (32l << 20));       // 16 MB
  bf16*  rk     = (bf16*) (ws + (48l << 20));       //  4 MB
  bf16*  av     = (bf16*) (ws + (52l << 20));       //  8 MB
  float* ao     = (float*)(ws + (60l << 20));       // 16 MB
  bf16*  catB   = (bf16*) (ws + (76l << 20));       // 16 MB  cat(mems,w)
  bf16*  qkvWB  = (bf16*) (ws + (92l << 20));       //  6 MB
  bf16*  rB     = (bf16*) (ws + (98l << 20));       //  4 MB
  bf16*  rWB    = (bf16*) (ws + (102l << 20));      //  2 MB
  bf16*  oWB    = (bf16*) (ws + (104l << 20));      //  2 MB

  // 0) one-shot f32 -> bf16 conversions
  cvt_kernel<<<4096, 256, 0, stream>>>(mems, catB);                 // 4M elems
  cvt_kernel<<<4096, 256, 0, stream>>>(w, catB + (long)MLEN * BSZ * DM);
  cvt_kernel<<<3072, 256, 0, stream>>>(qkvW, qkvWB);
  cvt_kernel<<<2048, 256, 0, stream>>>(r, rB);
  cvt_kernel<<<1024, 256, 0, stream>>>(rW, rWB);
  cvt_kernel<<<1024, 256, 0, stream>>>(oW, oWB);

  // 1) qkv projection: cat [8192 x 1024] x qkv_W^T -> q/k/vT heads
  gemm_bf16_kernel<<<dim3(8192 / 128, 3072 / 64), 256, 0, stream>>>(
      catB, qkvWB, 8192, 3072, DM, 0, head_q, head_k, vT);

  // 2) r projection -> rk bf16
  gemm_bf16_kernel<<<dim3(2048 / 128, 1024 / 64), 256, 0, stream>>>(
      rB, rWB, 2048, 1024, DM, 1, nullptr, rk, nullptr);

  // 3) attention (rel-shift + mask + online softmax + PV)
  size_t att_smem = 16 * 2048 * sizeof(float) + 16 * 32 * sizeof(bf16);
  attn_kernel<<<dim3(QLEN / 16, NH, BSZ), 32, att_smem, stream>>>(
      head_q, head_k, vT, rk, rwb, rrb, av);

  // 4) output projection: av [4096 x 1024] x o_W^T -> ao f32
  gemm_bf16_kernel<<<dim3(4096 / 128, 1024 / 64), 256, 0, stream>>>(
      av, oWB, 4096, 1024, DM, 2, ao, nullptr, nullptr);

  // 5) residual + layernorm
  ln_kernel<<<QLEN * BSZ, 256, 0, stream>>>(w, ao, ln_g, ln_b, (float*)d_out);
}
